// EncoderLayer_13666585936666
// MI455X (gfx1250) — matrix operand order repaired and checked by reading, not compiled, not run on hardware
//
#include <hip/hip_runtime.h>
#include <hip/hip_bf16.h>

// ---------------------------------------------------------------------------
// Types for CDNA5 WMMA (wave32): D(16x16 f32) = A(16x32 f16) * B(32x16 f16) + C
// ---------------------------------------------------------------------------
typedef __attribute__((ext_vector_type(16))) _Float16 v16h;
typedef __attribute__((ext_vector_type(8)))  _Float16 v8h;
typedef __attribute__((ext_vector_type(8)))  float    v8f;

#define HEADS 16
#define DH    64
#define SEQL  2048

__device__ __forceinline__ v8f wmma16x16x32(v16h a, v16h b, v8f c) {
  return __builtin_amdgcn_wmma_f32_16x16x32_f16(
      /*neg_a=*/false, a, /*neg_b=*/false, b,
      /*c_mod=*/(short)0, c, /*reuse_a=*/false, /*reuse_b=*/false);
}

__device__ __forceinline__ v16h frag_cat(v8h lo, v8h hi) {
  return __builtin_shufflevector(lo, hi, 0, 1, 2, 3, 4, 5, 6, 7,
                                 8, 9, 10, 11, 12, 13, 14, 15);
}

// A-fragment (16x32 f16): row `row` of a K-contiguous (row-major) matrix.
__device__ __forceinline__ v16h load_a_frag(const _Float16* base, int row, int ld, int lane) {
  const _Float16* p = base + (size_t)row * ld + (((lane >> 4) & 1) << 4);
  v8h lo = *reinterpret_cast<const v8h*>(p);
  v8h hi = *reinterpret_cast<const v8h*>(p + 8);
  return frag_cat(lo, hi);
}

// B-fragment (32x16 f16) from K-contiguous storage: element (k,n) at
// base[n*ld + koff + k].
__device__ __forceinline__ v16h load_b_frag(const _Float16* base, int n0, int ld, int koff, int lane) {
  const _Float16* p =
      base + (size_t)(n0 + (lane & 15)) * ld + koff + (((lane >> 4) & 1) << 4);
  v8h lo = *reinterpret_cast<const v8h*>(p);
  v8h hi = *reinterpret_cast<const v8h*>(p + 8);
  return frag_cat(lo, hi);
}

// ---------------------------------------------------------------------------
// Async global -> LDS copy (CDNA5 path, tracked by ASYNCcnt).
// ---------------------------------------------------------------------------
__device__ __forceinline__ void async_ld16(void* lds_dst, const void* gsrc) {
  unsigned off = (unsigned)(size_t)lds_dst;                  // low 32b = LDS addr
  unsigned long long ga = (unsigned long long)(size_t)gsrc;
  asm volatile("global_load_async_to_lds_b128 %0, %1, off"
               :: "v"(off), "v"(ga) : "memory");
}
__device__ __forceinline__ void wait_async0() {
  asm volatile("s_wait_asynccnt 0" ::: "memory");
}

// ---------------------------------------------------------------------------
// 16-lane (DPP-row) butterfly reductions: pure VALU, no LDS traffic.
// ---------------------------------------------------------------------------
template <int CTRL>
__device__ __forceinline__ float dpp_mov_f32(float v) {
  int i = __float_as_int(v);
  return __int_as_float(__builtin_amdgcn_update_dpp(i, i, CTRL, 0xF, 0xF, false));
}
__device__ __forceinline__ float row_max16(float v) {
  v = fmaxf(v, dpp_mov_f32<0xB1>(v));    // quad_perm xor1
  v = fmaxf(v, dpp_mov_f32<0x4E>(v));    // quad_perm xor2
  v = fmaxf(v, dpp_mov_f32<0x141>(v));   // row_half_mirror
  v = fmaxf(v, dpp_mov_f32<0x140>(v));   // row_mirror
  return v;
}
__device__ __forceinline__ float row_sum16(float v) {
  v += dpp_mov_f32<0xB1>(v);
  v += dpp_mov_f32<0x4E>(v);
  v += dpp_mov_f32<0x141>(v);
  v += dpp_mov_f32<0x140>(v);
  return v;
}

// ---------------------------------------------------------------------------
// Conversions.
// ---------------------------------------------------------------------------
__global__ void f32_to_f16_kernel(const float* __restrict__ s, _Float16* __restrict__ d, int n) {
  int i = (blockIdx.x * blockDim.x + threadIdx.x) * 4;
  if (i + 3 < n) {
#pragma unroll
    for (int t = 0; t < 4; ++t) d[i + t] = (_Float16)s[i + t];
  } else {
    for (int t = 0; t < 4 && i + t < n; ++t) d[i + t] = (_Float16)s[i + t];
  }
}

// Pack f32 [K,N] weights into WMMA-native f16 layout:
//   dst[((k/32)*N + n)*32 + (k%32)]
__global__ void pack_b_f16_kernel(const float* __restrict__ src, _Float16* __restrict__ dst,
                                  int K, int N) {
  const int t = blockIdx.x * blockDim.x + threadIdx.x;
  const int n = t % N;
  const int k0 = (t / N) * 8;
  if (k0 >= K) return;
  alignas(16) _Float16 tmp[8];
#pragma unroll
  for (int i = 0; i < 8; ++i) tmp[i] = (_Float16)src[(size_t)(k0 + i) * N + n];
  *reinterpret_cast<v8h*>(&dst[((size_t)(k0 >> 5) * N + n) * 32 + (k0 & 31)]) =
      *reinterpret_cast<const v8h*>(tmp);
}

__global__ void init_stats_kernel(float* stats) {
  if (threadIdx.x < 32) stats[threadIdx.x] = 0.0f;
}

// ---------------------------------------------------------------------------
// Tiled WMMA GEMM: C[M,N] = A[M,K](f16 row-major) * B(f16 packed) + bias.
// Block 128x128, 8 waves (each 32x64), K step 32, double-buffered LDS with
// async global->LDS staging overlapped with WMMA compute.
// W16T: scatter f16 output into head-transposed [B][HEADS][DH][SEQL] layout.
// ---------------------------------------------------------------------------
__device__ __forceinline__ void gemm_stage(const _Float16* __restrict__ A,
                                           const _Float16* __restrict__ Bp,
                                           int Kdim, int Ndim, int bm, int bn, int kt,
                                           _Float16 (*As)[40], _Float16 (*Bs)[40], int tid) {
  const int t8 = tid * 8;
  {  // A tile: 128 rows x 32 k
    const int ar = t8 >> 5, acol = t8 & 31;
#pragma unroll
    for (int half = 0; half < 2; ++half) {
      async_ld16(&As[ar + half * 64][acol],
                 A + (size_t)(bm + ar + half * 64) * Kdim + (size_t)kt * 32 + acol);
    }
  }
  {  // B tile (packed): 128 n x 32 k, contiguous in global
    const int nb = t8 >> 5, kk = t8 & 31;
    const _Float16* gb = Bp + ((size_t)kt * Ndim + bn) * 32;
#pragma unroll
    for (int half = 0; half < 2; ++half) {
      async_ld16(&Bs[nb + half * 64][kk],
                 gb + (size_t)(nb + half * 64) * 32 + kk);
    }
  }
}

template <int RELU, int W16, int W32, int W16T>
__global__ __launch_bounds__(256) void gemm_wmma_kernel(
    const _Float16* __restrict__ A, const _Float16* __restrict__ Bp,
    const float* __restrict__ bias, float* __restrict__ C32,
    _Float16* __restrict__ C16, _Float16* __restrict__ C16T,
    int Mdim, int Ndim, int Kdim) {
  __shared__ __align__(16) _Float16 As[2][128][40];
  __shared__ __align__(16) _Float16 Bs[2][128][40];

  const int lane = threadIdx.x & 31;
  const int wave = threadIdx.x >> 5;
  const int wm   = (wave >> 1) * 32;
  const int wn   = (wave & 1) * 64;
  const int bm   = blockIdx.y * 128;
  const int bn   = blockIdx.x * 128;

  v8f acc[2][4] = {};

  const int KT = Kdim >> 5;
  gemm_stage(A, Bp, Kdim, Ndim, bm, bn, 0, As[0], Bs[0], threadIdx.x);
  wait_async0();
  __syncthreads();

  for (int kt = 0; kt < KT; ++kt) {
    const int cur = kt & 1;
    if (kt + 1 < KT)
      gemm_stage(A, Bp, Kdim, Ndim, bm, bn, kt + 1, As[cur ^ 1], Bs[cur ^ 1], threadIdx.x);
    if (kt + 2 < KT) {  // warm L2 for the tile after next
      const int t8 = threadIdx.x * 8;
      __builtin_prefetch(A + (size_t)(bm + (t8 >> 5)) * Kdim + (size_t)(kt + 2) * 32 + (t8 & 31), 0, 3);
      __builtin_prefetch(Bp + ((size_t)(kt + 2) * Ndim + bn + (t8 >> 5)) * 32 + (t8 & 31), 0, 3);
    }

    v16h af[2], bf[4];
#pragma unroll
    for (int mt = 0; mt < 2; ++mt)
      af[mt] = load_a_frag(&As[cur][0][0], wm + mt * 16 + (lane & 15), 40, lane);
#pragma unroll
    for (int nt = 0; nt < 4; ++nt)
      bf[nt] = load_b_frag(&Bs[cur][0][0], wn + nt * 16, 40, 0, lane);
#pragma unroll
    for (int mt = 0; mt < 2; ++mt)
#pragma unroll
      for (int nt = 0; nt < 4; ++nt)
        acc[mt][nt] = wmma16x16x32(af[mt], bf[nt], acc[mt][nt]);

    wait_async0();          // next-buffer async copies done (overlapped with WMMA)
    __syncthreads();
  }

  const int hi = (lane >> 4) << 3;
#pragma unroll
  for (int mt = 0; mt < 2; ++mt) {
#pragma unroll
    for (int nt = 0; nt < 4; ++nt) {
      const int col = bn + wn + nt * 16 + (lane & 15);
      const float bv = bias[col];
#pragma unroll
      for (int r = 0; r < 8; ++r) {
        const int row = bm + wm + mt * 16 + r + hi;
        float v = acc[mt][nt][r] + bv;
        if (RELU) v = fmaxf(v, 0.0f);
        if (W32) C32[(size_t)row * Ndim + col] = v;
        if (W16) C16[(size_t)row * Ndim + col] = (_Float16)v;
        if (W16T) {  // head-transposed: [B][HEADS][DH][SEQL]
          const int hh = col >> 6, dd = col & 63;
          const int bb = row / SEQL, jj = row % SEQL;
          C16T[(((size_t)bb * HEADS + hh) * DH + dd) * SEQL + jj] = (_Float16)v;
        }
      }
    }
  }
}

// ---------------------------------------------------------------------------
// Flash-style multi-head attention.
// Grid: (L/128, B*HEADS). Block: 256 = 8 waves; each wave owns 16 Q rows.
// Q/K f16 in [B*L, D] (head h at column h*64); V f16 pre-transposed
// [B][HEADS][DH][L].  All chunk staging is async global->LDS.
// ---------------------------------------------------------------------------
__global__ __launch_bounds__(256) void attention_kernel(
    const _Float16* __restrict__ Qg, const _Float16* __restrict__ Kg,
    const _Float16* __restrict__ Vtg, float* __restrict__ Og, int L, int Dm) {
  const int lane = threadIdx.x & 31;
  const int wave = threadIdx.x >> 5;
  const int bh   = blockIdx.y;
  const int b    = bh >> 4;
  const int h    = bh & 15;
  const int q0   = blockIdx.x * 128 + wave * 16;

  const _Float16* Qb  = Qg + ((size_t)b * L) * Dm + (size_t)h * DH;
  const _Float16* Kb  = Kg + ((size_t)b * L) * Dm + (size_t)h * DH;
  const _Float16* Vtb = Vtg + ((size_t)bh * DH) * L;   // [DH][L] for this head

  __shared__ __align__(16) _Float16 Ks[32][72];     // K chunk row-major: Ks[key][dh]
  __shared__ __align__(16) _Float16 Vt[DH][40];     // V^T chunk: Vt[dh][key]
  __shared__ __align__(16) _Float16 Ps[8][16][40];  // per-wave P tile (16x32)

  const v16h qa0 = load_a_frag(Qb,      q0 + (lane & 15), Dm, lane);  // dh 0..31
  const v16h qa1 = load_a_frag(Qb + 32, q0 + (lane & 15), Dm, lane);  // dh 32..63

  v8f   o[4] = {};
  float mrow[8], lrow[8];
#pragma unroll
  for (int r = 0; r < 8; ++r) { mrow[r] = -1.0e30f; lrow[r] = 0.0f; }

  const float scl = 0.125f;  // 1/sqrt(64)
  const int   hi  = (lane >> 4) << 3;

  for (int j0 = 0; j0 < L; j0 += 32) {
    // Async staging: K chunk (32 keys x 64 dh) and V^T chunk (64 dh x 32 keys).
    {
      const int idx = threadIdx.x * 8;
      const int jr  = idx >> 6;   // key 0..31
      const int kc  = idx & 63;   // dh 0..56 step 8
      async_ld16(&Ks[jr][kc], Kb + (size_t)(j0 + jr) * Dm + kc);
      const int dhr = idx >> 5;   // dh 0..63
      const int jc  = idx & 31;   // key 0..24 step 8
      async_ld16(&Vt[dhr][jc], Vtb + (size_t)dhr * L + j0 + jc);
    }
    wait_async0();
    __syncthreads();

    // S = Q * K^T, two 16-key tiles, dh reduction via 2 chained WMMAs each.
    v8f s[2];
#pragma unroll
    for (int c = 0; c < 2; ++c) {
      v16h b0 = load_b_frag(&Ks[0][0], c * 16, 72, 0,  lane);  // dh 0..31
      v16h b1 = load_b_frag(&Ks[0][0], c * 16, 72, 32, lane);  // dh 32..63
      v8f z = {};
      z = wmma16x16x32(qa0, b0, z);
      s[c] = wmma16x16x32(qa1, b1, z);
    }

    // Online softmax (row r+hi; 32 columns across the 16-lane group, DPP reduce).
    float corr[8];
#pragma unroll
    for (int r = 0; r < 8; ++r) {
      const float v = row_max16(fmaxf(s[0][r], s[1][r]) * scl);
      const float mnew = fmaxf(mrow[r], v);
      corr[r] = __expf(mrow[r] - mnew);
      mrow[r] = mnew;

      const float p0 = __expf(s[0][r] * scl - mnew);
      const float p1 = __expf(s[1][r] * scl - mnew);
      lrow[r] = lrow[r] * corr[r] + row_sum16(p0 + p1);

      const int rr = r + hi;
      Ps[wave][rr][lane & 15]        = (_Float16)p0;
      Ps[wave][rr][16 + (lane & 15)] = (_Float16)p1;

      o[0][r] *= corr[r]; o[1][r] *= corr[r];
      o[2][r] *= corr[r]; o[3][r] *= corr[r];
    }

    // Cross-lane hand-off of P through per-wave LDS (in-order DS + wait).
    asm volatile("s_wait_dscnt 0" ::: "memory");
    const v16h pa = load_a_frag(&Ps[wave][0][0], lane & 15, 40, lane);

    // O += P * V  (dh = 4 column tiles; key reduction contiguous in Vt).
#pragma unroll
    for (int c2 = 0; c2 < 4; ++c2) {
      v16h vb = load_b_frag(&Vt[0][0], c2 * 16, 40, 0, lane);
      o[c2] = wmma16x16x32(pa, vb, o[c2]);
    }
    __syncthreads();
  }

#pragma unroll
  for (int c2 = 0; c2 < 4; ++c2) {
#pragma unroll
    for (int r = 0; r < 8; ++r) {
      const int row = q0 + r + hi;
      const int col = h * DH + c2 * 16 + (lane & 15);
      Og[((size_t)b * L + row) * Dm + col] = o[c2][r] / lrow[r];
    }
  }
}

// ---------------------------------------------------------------------------
// Joint (L,C) LayerNorm, two-phase.
// ---------------------------------------------------------------------------
__global__ void add_reduce_kernel(const float* __restrict__ a, const float* __restrict__ b,
                                  float* __restrict__ hbuf, float* __restrict__ stats,
                                  int per_batch) {
  const int bidx = blockIdx.y;
  const size_t base = (size_t)bidx * per_batch;
  const int i0 = (blockIdx.x * blockDim.x + threadIdx.x) * 8;
  float s1 = 0.0f, s2 = 0.0f;
#pragma unroll
  for (int t = 0; t < 8; ++t) {
    const size_t i = base + i0 + t;
    const float v = a[i] + b[i];
    hbuf[i] = v;
    s1 += v;
    s2 += v * v;
  }
  __shared__ float r1[256], r2[256];
  r1[threadIdx.x] = s1;
  r2[threadIdx.x] = s2;
  __syncthreads();
  for (int off = 128; off > 0; off >>= 1) {
    if ((int)threadIdx.x < off) {
      r1[threadIdx.x] += r1[threadIdx.x + off];
      r2[threadIdx.x] += r2[threadIdx.x + off];
    }
    __syncthreads();
  }
  if (threadIdx.x == 0) {
    atomicAdd(&stats[bidx * 2 + 0], r1[0]);
    atomicAdd(&stats[bidx * 2 + 1], r2[0]);
  }
}

__global__ void ln_norm_kernel(const float* __restrict__ hbuf, const float* __restrict__ stats,
                               float* __restrict__ o32, _Float16* __restrict__ o16,
                               int per_batch) {
  const int bidx = blockIdx.y;
  const float inv_n = 1.0f / (float)per_batch;
  const float mean = stats[bidx * 2 + 0] * inv_n;
  const float var  = stats[bidx * 2 + 1] * inv_n - mean * mean;
  const float rstd = rsqrtf(var + 1e-5f);
  const size_t base = (size_t)bidx * per_batch;
  const int i0 = (blockIdx.x * blockDim.x + threadIdx.x) * 4;
#pragma unroll
  for (int t = 0; t < 4; ++t) {
    const size_t i = base + i0 + t;
    const float v = (hbuf[i] - mean) * rstd;
    if (o32) o32[i] = v;
    if (o16) o16[i] = (_Float16)v;
  }
}

// ---------------------------------------------------------------------------
// Host-side orchestration (workspace ~110 MiB, dead-region reuse).
// ---------------------------------------------------------------------------
extern "C" void kernel_launch(void* const* d_in, const int* in_sizes, int n_in,
                              void* d_out, int out_size, void* d_ws, size_t ws_size,
                              hipStream_t stream) {
  (void)in_sizes; (void)n_in; (void)out_size; (void)ws_size;

  const int B = 2, L = 2048, D = 1024, FF = 4096;
  const int M = B * L;
  const int per_batch = L * D;

  const float* x  = (const float*)d_in[0];
  const float* wq = (const float*)d_in[1];
  const float* bq = (const float*)d_in[2];
  const float* wk = (const float*)d_in[3];
  const float* bk = (const float*)d_in[4];
  const float* wv = (const float*)d_in[5];
  const float* bv = (const float*)d_in[6];
  const float* w1 = (const float*)d_in[7];
  const float* b1 = (const float*)d_in[8];
  const float* w2 = (const float*)d_in[9];
  const float* b2 = (const float*)d_in[10];

  const size_t MB = 1ull << 20;
  char* w = (char*)d_ws;
  _Float16* xh     = (_Float16*)(w + 0 * MB);
  _Float16* wqp    = (_Float16*)(w + 8 * MB);
  _Float16* wkp    = (_Float16*)(w + 10 * MB);
  _Float16* wvp    = (_Float16*)(w + 12 * MB);
  _Float16* w1p    = (_Float16*)(w + 14 * MB);
  _Float16* w2p    = (_Float16*)(w + 22 * MB);
  _Float16* qh     = (_Float16*)(w + 30 * MB);
  _Float16* kh     = (_Float16*)(w + 38 * MB);
  _Float16* vt     = (_Float16*)(w + 46 * MB);  // head-transposed V
  float*    attn   = (float*)   (w + 54 * MB);
  float*    h_raw  = (float*)   (w + 70 * MB);
  float*    h_norm = (float*)   (w + 86 * MB);
  _Float16* h16    = (_Float16*)(w + 102 * MB);
  _Float16* ff1h   = (_Float16*)(w + 30 * MB);  // reuse q/k/vt (dead after attention)
  float*    ff2    = (float*)   (w + 62 * MB);  // reuse attn/h_raw tails (dead)
  float*    g_raw  = (float*)   (w + 0 * MB);   // reuse xh/weights (dead)
  float*    stats  = (float*)   (w + 110 * MB);

  init_stats_kernel<<<1, 32, 0, stream>>>(stats);

  f32_to_f16_kernel<<<(M * D) / 1024, 256, 0, stream>>>(x, xh, M * D);
  pack_b_f16_kernel<<<(D / 8) * D / 256, 256, 0, stream>>>(wq, wqp, D, D);
  pack_b_f16_kernel<<<(D / 8) * D / 256, 256, 0, stream>>>(wk, wkp, D, D);
  pack_b_f16_kernel<<<(D / 8) * D / 256, 256, 0, stream>>>(wv, wvp, D, D);
  pack_b_f16_kernel<<<(D / 8) * FF / 256, 256, 0, stream>>>(w1, w1p, D, FF);
  pack_b_f16_kernel<<<(FF / 8) * D / 256, 256, 0, stream>>>(w2, w2p, FF, D);

  dim3 gQKV(D / 128, M / 128);
  gemm_wmma_kernel<0, 1, 0, 0><<<gQKV, 256, 0, stream>>>(xh, wqp, bq, nullptr, qh, nullptr, M, D, D);
  gemm_wmma_kernel<0, 1, 0, 0><<<gQKV, 256, 0, stream>>>(xh, wkp, bk, nullptr, kh, nullptr, M, D, D);
  gemm_wmma_kernel<0, 0, 0, 1><<<gQKV, 256, 0, stream>>>(xh, wvp, bv, nullptr, nullptr, vt, M, D, D);

  attention_kernel<<<dim3(L / 128, B * HEADS), 256, 0, stream>>>(qh, kh, vt, attn, L, D);

  add_reduce_kernel<<<dim3(per_batch / (256 * 8), B), 256, 0, stream>>>(x, attn, h_raw, stats, per_batch);
  ln_norm_kernel<<<dim3(per_batch / (256 * 4), B), 256, 0, stream>>>(h_raw, stats, h_norm, h16, per_batch);

  gemm_wmma_kernel<1, 1, 0, 0><<<dim3(FF / 128, M / 128), 256, 0, stream>>>(h16, w1p, b1, nullptr, ff1h, nullptr, M, FF, D);
  gemm_wmma_kernel<0, 0, 1, 0><<<dim3(D / 128, M / 128), 256, 0, stream>>>(ff1h, w2p, b2, ff2, nullptr, nullptr, M, D, FF);

  add_reduce_kernel<<<dim3(per_batch / (256 * 8), B), 256, 0, stream>>>(h_norm, ff2, g_raw, stats + 8, per_batch);
  ln_norm_kernel<<<dim3(per_batch / (256 * 4), B), 256, 0, stream>>>(g_raw, stats + 8, (float*)d_out, nullptr, per_batch);
}
